// MultiHeadAttention_58385785422069
// MI455X (gfx1250) — matrix operand order repaired and checked by reading, not compiled, not run on hardware
//
#include <hip/hip_runtime.h>
#include <hip/hip_bf16.h>

// ---------------------------------------------------------------------------
// MultiHeadAttention forward for MI455X (gfx1250, wave32, WMMA + TDM).
// Shapes: B=4, L=1024, DM=1024, H=16, DK=DV=64.
// Outputs (concatenated in d_out): out [B,L,DM] f32, attn [B,H,L,L] f32.
// GEMMs on v_wmma_f32_16x16x32_f16; weight B-tiles staged into LDS by the
// Tensor Data Mover (tensor_load_to_lds + s_wait_tensorcnt).
// ---------------------------------------------------------------------------

typedef __attribute__((ext_vector_type(16))) _Float16 v16h;
typedef __attribute__((ext_vector_type(8)))  _Float16 v8h;
typedef __attribute__((ext_vector_type(8)))  float    v8f;

#define DEVINL __device__ __forceinline__

constexpr int Bc  = 4;
constexpr int Lc  = 1024;
constexpr int DMc = 1024;
constexpr int Hc  = 16;
constexpr int DKc = 64;
constexpr int DVc = 64;

#if defined(__has_builtin)
#if __has_builtin(__builtin_amdgcn_tensor_load_to_lds) && \
    __has_builtin(__builtin_amdgcn_s_wait_tensorcnt)
#define ATHENA_HAVE_TDM 1
#endif
#endif
#ifndef ATHENA_HAVE_TDM
#define ATHENA_HAVE_TDM 0
#endif

DEVINL int lane_id() { return (int)(threadIdx.x & 31u); }

DEVINL v8f wmma_16x16x32(v16h a, v16h b, v8f c) {
  // (neg_a, A, neg_b, B, c_mod, C, reuse_a, reuse_b)
  return __builtin_amdgcn_wmma_f32_16x16x32_f16(false, a, false, b, (short)0, c,
                                                false, false);
}

// --- A fragment (16x32, MxK) ------------------------------------------------
// Lanes 0-15 row M=lane, K 0..15; lanes 16-31 K 16..31 (matches B).
// src is row-major [row][k] with leading dim ld.
DEVINL v16h load_a_f32(const float* __restrict__ src, int row0, int k0, int ld) {
  const int lane = lane_id();
  const float* p = src + (long)(row0 + (lane & 15)) * ld + k0 + ((lane >> 4) << 4);
  v16h a;
#pragma unroll
  for (int i = 0; i < 8; ++i) {
    a[i]     = (_Float16)p[i];
    a[i + 8] = (_Float16)p[i + 8];
  }
  return a;
}

DEVINL v16h load_a_f16(const _Float16* __restrict__ src, int row0, int k0, int ld) {
  const int lane = lane_id();
  const _Float16* p = src + (long)(row0 + (lane & 15)) * ld + k0 + ((lane >> 4) << 4);
  v8h lo = *(const v8h*)p;          // 16B aligned
  v8h hi = *(const v8h*)(p + 8);    // 16B aligned
  v16h a;
#pragma unroll
  for (int i = 0; i < 8; ++i) { a[i] = lo[i]; a[i + 8] = hi[i]; }
  return a;
}

// --- B fragment (32x16, KxN), sourced as B[k][n] = srcT[n][k] ---------------
// Lane = column N; lanes 0-15 hold K 0..15, lanes 16-31 hold K 16..31
// (contiguous k = 16h + s, same order as A).  srcT is row-major [n][k] with leading dim ld.
DEVINL v16h load_bt_f32(const float* __restrict__ srcT, int n0, int k0, int ld) {
  const int lane = lane_id();
  const float* p = srcT + (long)(n0 + (lane & 15)) * ld + k0 + ((lane >> 4) << 4);
  v16h b;
#pragma unroll
  for (int i = 0; i < 16; ++i) b[i] = (_Float16)p[i];
  return b;
}

DEVINL v16h load_bt_f16(const _Float16* __restrict__ srcT, int n0, int k0, int ld) {
  const int lane = lane_id();
  return *(const v16h*)(srcT + (long)(n0 + (lane & 15)) * ld + k0 +
                        ((lane >> 4) << 4));  // 32B aligned
}

// B fragment from an LDS tile (row-major [n_local][ld] f32) -> ds_load + cvt.
DEVINL v16h load_bt_f32_lds(const float* s, int n0, int k0, int ld) {
  const int lane = lane_id();
  const float* p = s + (n0 + (lane & 15)) * ld + k0 + ((lane >> 4) << 4);
  v16h b;
#pragma unroll
  for (int i = 0; i < 16; ++i) b[i] = (_Float16)p[i];
  return b;
}

// ---------------------------------------------------------------------------
// Tensor Data Mover: stage a 2D f32 tile (tile_y rows x tile_x elems) into
// LDS offset lds_off.  D# built per ISA ch.8 (group0: count/lds/global/type,
// group1: data_size + dims + strides).  Issued per-wave (EXEC ignored).
// ---------------------------------------------------------------------------
#if ATHENA_HAVE_TDM
typedef __attribute__((ext_vector_type(4))) unsigned int u32x4;
typedef __attribute__((ext_vector_type(8))) int i32x8;
typedef __attribute__((ext_vector_type(4))) int i32x4;

DEVINL void tdm_load_2d_f32(unsigned lds_off, const float* g,
                            unsigned tile_x, unsigned tile_y,
                            unsigned tensor_x, unsigned tensor_y,
                            unsigned row_stride) {
  const unsigned long long ga = (unsigned long long)g;
  u32x4 g0;
  g0[0] = 1u;                                   // count=1, user descriptor
  g0[1] = lds_off;                              // lds_addr (bytes)
  g0[2] = (unsigned)ga;                         // global_addr[31:0]
  g0[3] = (unsigned)((ga >> 32) & 0x01FFFFFFull) | (2u << 30);  // addr[56:32]|type=2
  i32x8 g1;
  g1[0] = (int)(2u << 16);                      // wg_mask=0, data_size=2 (4B)
  g1[1] = (int)((tensor_x & 0xFFFFu) << 16);    // tensor_dim0[15:0]
  g1[2] = (int)((tensor_x >> 16) | ((tensor_y & 0xFFFFu) << 16));
  g1[3] = (int)((tensor_y >> 16) | (tile_x << 16));   // tile_dim0
  g1[4] = (int)(tile_y & 0xFFFFu);              // tile_dim1 (tile_dim2=0)
  g1[5] = (int)row_stride;                      // tensor_dim0_stride[31:0]
  g1[6] = 0;                                    // stride hi / dim1_stride lo
  g1[7] = 0;
  i32x4 z4 = {};
#if __clang_major__ >= 23
  i32x8 z8 = {};
  __builtin_amdgcn_tensor_load_to_lds(g0, g1, z4, z4, z8, 0);
#else
  __builtin_amdgcn_tensor_load_to_lds(g0, g1, z4, z4, 0);
#endif
}
#endif  // ATHENA_HAVE_TDM

// Stage W[n0b..n0b+64)[k0..k0+32) (row-major, ld elems) into wtile (LDS off 0).
// Caller must __syncthreads() before (LDS reuse); this syncs after.
DEVINL void stage_b_tile(const float* __restrict__ W, int n0b, int k0, int ld,
                         float* wtile) {
#if ATHENA_HAVE_TDM
  if (threadIdx.x < 32) {
    tdm_load_2d_f32(0u, W + (long)n0b * ld + k0, 32u, 64u,
                    (unsigned)ld, 1024u, (unsigned)ld);
    __builtin_amdgcn_s_wait_tensorcnt(0);
  }
#else
  for (int idx = (int)threadIdx.x; idx < 64 * 32; idx += 128) {
    const int nloc = idx >> 5, kloc = idx & 31;
    wtile[idx] = W[(long)(n0b + nloc) * ld + k0 + kloc];
  }
#endif
  __syncthreads();
}

// ---------------------------------------------------------------------------
// Kernel 1: QKV projections.  C[4096,1024] = X @ W^T + bias, stored f16.
// gridDim.z selects q/k/v.  Q,K stored head-major [bh][l][dk]; V stored
// transposed [bh][dv][l].  Block = 128 (4 waves); block tile 64x64; the W
// B-tile (64n x 32k) is TDM-staged into LDS once per k-step for all 4 waves.
// ---------------------------------------------------------------------------
__global__ void qkv_proj_kernel(const float* __restrict__ q,
                                const float* __restrict__ k,
                                const float* __restrict__ v,
                                const float* __restrict__ w_q, const float* __restrict__ b_q,
                                const float* __restrict__ w_k, const float* __restrict__ b_k,
                                const float* __restrict__ w_v, const float* __restrict__ b_v,
                                _Float16* __restrict__ Qh,
                                _Float16* __restrict__ Kh,
                                _Float16* __restrict__ Vt) {
  __shared__ float wtile[64 * 32];  // only LDS object -> offset 0 for TDM

  const int which = blockIdx.z;
  const float* X;
  const float* W;
  const float* bias;
  if (which == 0)      { X = q; W = w_q; bias = b_q; }
  else if (which == 1) { X = k; W = w_k; bias = b_k; }
  else                 { X = v; W = w_v; bias = b_v; }

  const int m0  = blockIdx.x * 64;
  const int wv  = (int)(threadIdx.x >> 5);
  const int n0b = blockIdx.y * 64;
  const int n0  = n0b + wv * 16;

  v8f acc[4] = {};
  for (int k0 = 0; k0 < DMc; k0 += 32) {
    __syncthreads();                       // previous-iteration LDS reads done
    stage_b_tile(W, n0b, k0, DMc, wtile);  // TDM + s_wait_tensorcnt + barrier
    v16h bf = load_bt_f32_lds(wtile, wv * 16, 0, 32);
#pragma unroll
    for (int i = 0; i < 4; ++i) {
      v16h af = load_a_f32(X, m0 + i * 16, k0, DMc);
      acc[i] = wmma_16x16x32(af, bf, acc[i]);
    }
  }

  // C layout: lane&15 = N, (lane>>4)*8 + r = M within tile.
  const int lane = lane_id();
  const int n  = n0 + (lane & 15);
  const int mb = (lane >> 4) << 3;
  const float bn = bias[n];
  const int h = n >> 6, d = n & 63;
#pragma unroll
  for (int i = 0; i < 4; ++i) {
#pragma unroll
    for (int r = 0; r < 8; ++r) {
      const int m  = m0 + i * 16 + mb + r;
      const int bb = m >> 10, l = m & 1023;
      const float val = acc[i][r] + bn;
      if (which == 2) {
        Vt[(long)((bb * Hc + h) * DVc + d) * Lc + l] = (_Float16)val;
      } else {
        _Float16* dst = (which == 0) ? Qh : Kh;
        dst[(long)((bb * Hc + h) * Lc + l) * DKc + d] = (_Float16)val;
      }
    }
  }
}

// ---------------------------------------------------------------------------
// Kernel 2: scores = (Qh Kh^T)/8 * gate, masked -> -inf.  Writes logits into
// the attn region of d_out.  Block = 128 (4 waves); wave does one 16-row
// q-tile x 4 k-tiles; 2 WMMA per tile.  grid = (64 qtiles, 4 kchunks, 64 bh)
// ---------------------------------------------------------------------------
__global__ void scores_kernel(const _Float16* __restrict__ Qh,
                              const _Float16* __restrict__ Kh,
                              const float* __restrict__ k_gate,
                              const int* __restrict__ mask,
                              float* __restrict__ attn) {
  const int bh = blockIdx.z;
  const int qt = blockIdx.x;
  const int wv = (int)(threadIdx.x >> 5);
  const int kt0 = blockIdx.y * 16 + wv * 4;

  const _Float16* Qbh = Qh + (long)bh * Lc * DKc;
  const _Float16* Kbh = Kh + (long)bh * Lc * DKc;
  const float* gbh = k_gate + (long)bh * Lc * Lc;
  const int*   msk = mask + (long)(bh >> 4) * Lc * Lc;
  float*       abh = attn + (long)bh * Lc * Lc;

  const int lane = lane_id();
  const int nn = lane & 15;
  const int mbase = qt * 16 + ((lane >> 4) << 3);

  // Pull the epilogue gate/mask stream toward GL2 early (global_prefetch_b8).
  __builtin_prefetch(gbh + (long)mbase * Lc + kt0 * 16 + nn, 0, 3);
  __builtin_prefetch(msk + (long)mbase * Lc + kt0 * 16 + nn, 0, 3);

  const v16h a0 = load_a_f16(Qbh, qt * 16, 0, DKc);
  const v16h a1 = load_a_f16(Qbh, qt * 16, 32, DKc);

#pragma unroll
  for (int j = 0; j < 4; ++j) {
    const int kt = kt0 + j;
    v16h b0 = load_bt_f16(Kbh, kt * 16, 0, DKc);
    v16h b1 = load_bt_f16(Kbh, kt * 16, 32, DKc);
    v8f c = {};
    c = wmma_16x16x32(a0, b0, c);
    c = wmma_16x16x32(a1, b1, c);
    const int kc = kt * 16 + nn;
#pragma unroll
    for (int r = 0; r < 8; ++r) {
      const int qr = mbase + r;
      float s = c[r] * 0.125f * gbh[(long)qr * Lc + kc];
      if (msk[(long)qr * Lc + kc] > 0) s = -__builtin_inff();
      abh[(long)qr * Lc + kc] = s;
    }
  }
}

// ---------------------------------------------------------------------------
// Kernel 3: in-place row softmax over Lk=1024.  One 256-thread block per row.
// ---------------------------------------------------------------------------
__global__ void softmax_kernel(float* __restrict__ attn) {
  __shared__ float red[256];
  float* p = attn + (long)blockIdx.x * Lc;
  const int t = (int)threadIdx.x;

  float x0 = p[t], x1 = p[t + 256], x2 = p[t + 512], x3 = p[t + 768];
  float mx = fmaxf(fmaxf(x0, x1), fmaxf(x2, x3));
  red[t] = mx;
  __syncthreads();
  for (int s = 128; s > 0; s >>= 1) {
    if (t < s) red[t] = fmaxf(red[t], red[t + s]);
    __syncthreads();
  }
  mx = red[0];
  __syncthreads();

  float e0 = __expf(x0 - mx), e1 = __expf(x1 - mx);
  float e2 = __expf(x2 - mx), e3 = __expf(x3 - mx);
  red[t] = e0 + e1 + e2 + e3;
  __syncthreads();
  for (int s = 128; s > 0; s >>= 1) {
    if (t < s) red[t] += red[t + s];
    __syncthreads();
  }
  const float inv = 1.0f / red[0];
  p[t]       = e0 * inv;
  p[t + 256] = e1 * inv;
  p[t + 512] = e2 * inv;
  p[t + 768] = e3 * inv;
}

// ---------------------------------------------------------------------------
// Kernel 4: O = attn @ V.  Per wave: 16(q) x 64(dv) tile, 32 WMMA k-steps.
// attn read as f32 -> f16 A-fragments; V read from transposed f16 Vt.
// grid = (16 qgroups, 64 bh); block = 128 (4 waves, one qtile each)
// ---------------------------------------------------------------------------
__global__ void pv_kernel(const float* __restrict__ attn,
                          const _Float16* __restrict__ Vt,
                          _Float16* __restrict__ O) {
  const int bh = blockIdx.y;
  const int wv = (int)(threadIdx.x >> 5);
  const int qt = blockIdx.x * 4 + wv;

  const float*    abh = attn + (long)bh * Lc * Lc;
  const _Float16* vbh = Vt + (long)bh * DVc * Lc;

  v8f acc[4] = {};
  for (int k0 = 0; k0 < Lc; k0 += 32) {
    v16h af = load_a_f32(abh, qt * 16, k0, Lc);
#pragma unroll
    for (int nt = 0; nt < 4; ++nt) {
      v16h bf = load_bt_f16(vbh, nt * 16, k0, Lc);
      acc[nt] = wmma_16x16x32(af, bf, acc[nt]);
    }
  }

  const int lane = lane_id();
  const int nn = lane & 15;
  const int mb = (lane >> 4) << 3;
  const int b = bh >> 4, h = bh & 15;
#pragma unroll
  for (int nt = 0; nt < 4; ++nt) {
    const int d = nt * 16 + nn;
#pragma unroll
    for (int r = 0; r < 8; ++r) {
      const int l = qt * 16 + mb + r;
      O[(long)(b * Lc + l) * (Hc * DVc) + h * DVc + d] = (_Float16)acc[nt][r];
    }
  }
}

// ---------------------------------------------------------------------------
// Kernel 5: xres = O @ w_fc^T + b_fc + residual(q).  Same tiling as kernel 1;
// w_fc B-tiles TDM-staged through LDS.
// ---------------------------------------------------------------------------
__global__ void fc_kernel(const _Float16* __restrict__ O,
                          const float* __restrict__ w_fc,
                          const float* __restrict__ b_fc,
                          const float* __restrict__ qin,
                          float* __restrict__ xres) {
  __shared__ float wtile[64 * 32];  // only LDS object -> offset 0 for TDM

  const int m0  = blockIdx.x * 64;
  const int wv  = (int)(threadIdx.x >> 5);
  const int n0b = blockIdx.y * 64;
  const int n0  = n0b + wv * 16;

  v8f acc[4] = {};
  for (int k0 = 0; k0 < Hc * DVc; k0 += 32) {
    __syncthreads();
    stage_b_tile(w_fc, n0b, k0, Hc * DVc, wtile);
    v16h bf = load_bt_f32_lds(wtile, wv * 16, 0, 32);
#pragma unroll
    for (int i = 0; i < 4; ++i) {
      v16h af = load_a_f16(O, m0 + i * 16, k0, Hc * DVc);
      acc[i] = wmma_16x16x32(af, bf, acc[i]);
    }
  }

  const int lane = lane_id();
  const int n  = n0 + (lane & 15);
  const int mb = (lane >> 4) << 3;
  const float bn = b_fc[n];
#pragma unroll
  for (int i = 0; i < 4; ++i) {
#pragma unroll
    for (int r = 0; r < 8; ++r) {
      const int m = m0 + i * 16 + mb + r;
      xres[(long)m * DMc + n] = acc[i][r] + bn + qin[(long)m * DMc + n];
    }
  }
}

// ---------------------------------------------------------------------------
// Kernel 6: LayerNorm over d_model -> final out.  One block per row.
// ---------------------------------------------------------------------------
__global__ void ln_kernel(const float* __restrict__ x,
                          const float* __restrict__ g,
                          const float* __restrict__ b,
                          float* __restrict__ out) {
  __shared__ float red[256];
  const float* p = x + (long)blockIdx.x * DMc;
  float* o = out + (long)blockIdx.x * DMc;
  const int t = (int)threadIdx.x;

  float x0 = p[t], x1 = p[t + 256], x2 = p[t + 512], x3 = p[t + 768];
  red[t] = x0 + x1 + x2 + x3;
  __syncthreads();
  for (int s = 128; s > 0; s >>= 1) {
    if (t < s) red[t] += red[t + s];
    __syncthreads();
  }
  const float mu = red[0] * (1.0f / DMc);
  __syncthreads();

  float d0 = x0 - mu, d1 = x1 - mu, d2 = x2 - mu, d3 = x3 - mu;
  red[t] = d0 * d0 + d1 * d1 + d2 * d2 + d3 * d3;
  __syncthreads();
  for (int s = 128; s > 0; s >>= 1) {
    if (t < s) red[t] += red[t + s];
    __syncthreads();
  }
  const float rstd = rsqrtf(red[0] * (1.0f / DMc) + 1e-5f);
  o[t]       = d0 * rstd * g[t]       + b[t];
  o[t + 256] = d1 * rstd * g[t + 256] + b[t + 256];
  o[t + 512] = d2 * rstd * g[t + 512] + b[t + 512];
  o[t + 768] = d3 * rstd * g[t + 768] + b[t + 768];
}

// ---------------------------------------------------------------------------
extern "C" void kernel_launch(void* const* d_in, const int* in_sizes, int n_in,
                              void* d_out, int out_size, void* d_ws, size_t ws_size,
                              hipStream_t stream) {
  (void)in_sizes; (void)n_in; (void)out_size; (void)ws_size;

  const float* q      = (const float*)d_in[0];
  const float* k      = (const float*)d_in[1];
  const float* v      = (const float*)d_in[2];
  const int*   mask   = (const int*)d_in[3];
  const float* k_gate = (const float*)d_in[4];
  const float* w_q    = (const float*)d_in[5];
  const float* b_q    = (const float*)d_in[6];
  const float* w_k    = (const float*)d_in[7];
  const float* b_k    = (const float*)d_in[8];
  const float* w_v    = (const float*)d_in[9];
  const float* b_v    = (const float*)d_in[10];
  const float* w_fc   = (const float*)d_in[11];
  const float* b_fc   = (const float*)d_in[12];
  const float* ln_g   = (const float*)d_in[13];
  const float* ln_b   = (const float*)d_in[14];

  float* out  = (float*)d_out;                       // [B,L,DM]
  float* attn = out + (size_t)Bc * Lc * DMc;         // [B,H,L,L]

  // Workspace layout (48 MiB total; lives in the 192 MB L2):
  char* ws = (char*)d_ws;
  _Float16* Qh = (_Float16*)(ws);                        //  8 MiB [bh][l][dk]
  _Float16* Kh = (_Float16*)(ws + (size_t)(8u  << 20));  //  8 MiB [bh][l][dk]
  _Float16* Vt = (_Float16*)(ws + (size_t)(16u << 20));  //  8 MiB [bh][dv][l]
  _Float16* O  = (_Float16*)(ws + (size_t)(24u << 20));  //  8 MiB [m][h*dv+d]
  float*    xr = (float*)   (ws + (size_t)(32u << 20));  // 16 MiB residual sum

  qkv_proj_kernel<<<dim3(64, 16, 3), 128, 0, stream>>>(
      q, k, v, w_q, b_q, w_k, b_k, w_v, b_v, Qh, Kh, Vt);

  scores_kernel<<<dim3(64, 4, 64), 128, 0, stream>>>(Qh, Kh, k_gate, mask, attn);

  softmax_kernel<<<dim3(Bc * Hc * Lc), 256, 0, stream>>>(attn);

  pv_kernel<<<dim3(16, 64), 128, 0, stream>>>(attn, Vt, O);

  fc_kernel<<<dim3(64, 16), 128, 0, stream>>>(O, w_fc, b_fc, q, xr);

  ln_kernel<<<dim3(Bc * Lc), 256, 0, stream>>>(xr, ln_g, ln_b, out);
}
